// GCN_69956427317977
// MI455X (gfx1250) — matrix operand order repaired and checked by reading, not compiled, not run on hardware
//
#include <hip/hip_runtime.h>
#include <hip/hip_bf16.h>

typedef __attribute__((ext_vector_type(16))) __bf16 v16bf;
typedef __attribute__((ext_vector_type(8)))  float  v8f;

// ---------------------------------------------------------------------------
// utility
// ---------------------------------------------------------------------------
__device__ __forceinline__ void atomic_add_f32(float* p, float v) {
    __hip_atomic_fetch_add(p, v, __ATOMIC_RELAXED, __HIP_MEMORY_SCOPE_AGENT);
}

__global__ void zero_f32_kernel(float* __restrict__ p, long long n) {
    long long i = (long long)blockIdx.x * blockDim.x + threadIdx.x;
    if (i < n) p[i] = 0.0f;
}

// ---------------------------------------------------------------------------
// degree / dinv
// ---------------------------------------------------------------------------
__global__ void deg_kernel(const int* __restrict__ dst, float* __restrict__ deg, int E) {
    int e = blockIdx.x * blockDim.x + threadIdx.x;
    if (e < E) atomic_add_f32(&deg[dst[e]], 1.0f);
}

__global__ void dinv_kernel(float* __restrict__ deg, int N) {
    int i = blockIdx.x * blockDim.x + threadIdx.x;
    if (i < N) {
        float d = deg[i] + 1.0f;  // +1 self loop; always > 0
        deg[i] = __frsqrt_rn(d);  // store dinv in place
    }
}

// ---------------------------------------------------------------------------
// Pack W (f32, K x Nout row-major) into bf16 B-fragment layout:
//   Wp[ ((kt*Nout + n)*2 + half)*16 + j ] = bf16( W[ (kt*32 + half*16 + j)*Nout + n ] )
// so a lane (column n = 16*tile + (lane&15), half = lane>>4) reads its whole
// 32-byte fragment for K-step kt with one contiguous v16bf load.
// ---------------------------------------------------------------------------
__global__ void pack_W_kernel(const float* __restrict__ W, __bf16* __restrict__ Wp,
                              int K, int Nout) {
    int tid = blockIdx.x * blockDim.x + threadIdx.x;
    if (tid >= K * Nout) return;
    int j    = tid & 15;
    int half = (tid >> 4) & 1;
    int n    = (tid >> 5) % Nout;
    int kt   = (tid >> 5) / Nout;
    Wp[tid] = (__bf16)W[(size_t)(kt * 32 + half * 16 + j) * Nout + n];
}

// ---------------------------------------------------------------------------
// GEMM: Out[nrows x NOUT] = A[nrows x K] @ W[K x NOUT]
// bf16 WMMA, f32 accumulate. One wave computes a 16x64 tile (4 accumulators);
// A fragment loaded once per K-step and reused by 4 WMMAs. K-loop fully
// unrolled (templated) -> dense v_wmma stream.
// ---------------------------------------------------------------------------
template <int K, int NOUT>
__global__ __launch_bounds__(256)
void gemm_wmma_bf16(const float* __restrict__ A, const __bf16* __restrict__ Wp,
                    float* __restrict__ Out, int nrows) {
    constexpr int TN4 = NOUT / 64;                 // 64-column groups
    const int lane   = threadIdx.x & 31;
    const int wave   = blockIdx.x * 8 + (threadIdx.x >> 5);
    const int tilesM = (nrows + 15) >> 4;
    if (wave >= tilesM * TN4) return;

    const int tm   = wave / TN4;
    const int tn   = (wave % TN4) * 64;
    const int half = lane >> 4;                    // 0: lanes 0-15, 1: lanes 16-31
    const int m    = lane & 15;

    const int  arow   = tm * 16 + m;
    const float* Arow = A + (size_t)(arow < nrows ? arow : 0) * K;

    v8f acc[4] = {};

#pragma unroll
    for (int k0 = 0; k0 < K; k0 += 32) {
        // A fragment: elt j -> K = k0 + half*16 + j
        v16bf a;
        const float* a0 = Arow + k0 + half * 16;
#pragma unroll
        for (int j = 0; j < 8; ++j) {
            a[j]     = (__bf16)a0[j];
            a[8 + j] = (__bf16)a0[8 + j];
        }
#pragma unroll
        for (int t = 0; t < 4; ++t) {
            const v16bf b = *(const v16bf*)(
                Wp + ((size_t)(k0 >> 5) * NOUT + (tn + t * 16 + m)) * 32 + half * 16);
            acc[t] = __builtin_amdgcn_wmma_f32_16x16x32_bf16(
                /*neg_a=*/false, a, /*neg_b=*/false, b,
                /*c_mod=*/(short)0, acc[t], /*reuse_a=*/false, /*reuse_b=*/false);
        }
    }

    // D layout: VGPR r -> row tm*16 + half*8 + r, col = 16*tile + (lane&15)
    const int rbase = tm * 16 + half * 8;
    if (rbase + 8 <= nrows) {   // fast path: whole tile in range (always, N%16==0)
#pragma unroll
        for (int t = 0; t < 4; ++t) {
            float* op = Out + (size_t)rbase * NOUT + tn + t * 16 + m;
#pragma unroll
            for (int r = 0; r < 8; ++r) op[(size_t)r * NOUT] = acc[t][r];
        }
    } else {
#pragma unroll
        for (int t = 0; t < 4; ++t)
#pragma unroll
            for (int r = 0; r < 8; ++r)
                if (rbase + r < nrows)
                    Out[(size_t)(rbase + r) * NOUT + tn + t * 16 + m] = acc[t][r];
    }
}

// ---------------------------------------------------------------------------
// edge scatter: agg[dst] += lin[src] * dinv[src]*dinv[dst]   (float4 per thread)
// ---------------------------------------------------------------------------
__global__ void scatter_edges(const float* __restrict__ lin, const float* __restrict__ dinv,
                              const int* __restrict__ src, const int* __restrict__ dst,
                              float* __restrict__ agg, int E, int C) {
    const int c4 = C >> 2;
    long long tid = (long long)blockIdx.x * blockDim.x + threadIdx.x;
    if (tid >= (long long)E * c4) return;
    int e = (int)(tid / c4);
    int c = (int)(tid % c4) * 4;
    int s = src[e], d = dst[e];
    float nrm = dinv[s] * dinv[d];
    const float4 v = *(const float4*)(lin + (size_t)s * C + c);
    float* o = agg + (size_t)d * C + c;
    atomic_add_f32(o + 0, v.x * nrm);
    atomic_add_f32(o + 1, v.y * nrm);
    atomic_add_f32(o + 2, v.z * nrm);
    atomic_add_f32(o + 3, v.w * nrm);
}

// ---------------------------------------------------------------------------
// self-loop + bias + optional relu (in place on agg)
// h[i,c] = relu?( agg[i,c] + lin[i,c]*dinv[i]^2 + b[c] )
// ---------------------------------------------------------------------------
__global__ void fuse_kernel(float* __restrict__ agg, const float* __restrict__ lin,
                            const float* __restrict__ dinv, const float* __restrict__ bias,
                            int N, int C, int do_relu) {
    long long tid = (long long)blockIdx.x * blockDim.x + threadIdx.x;
    if (tid >= (long long)N * C) return;
    int i = (int)(tid / C);
    int c = (int)(tid % C);
    float di = dinv[i];
    float v = agg[tid] + lin[tid] * di * di + bias[c];
    if (do_relu) v = fmaxf(v, 0.0f);
    agg[tid] = v;
}

// ---------------------------------------------------------------------------
// pooling
// ---------------------------------------------------------------------------
__global__ void cnt_kernel(const int* __restrict__ batch, float* __restrict__ cnt, int N) {
    int i = blockIdx.x * blockDim.x + threadIdx.x;
    if (i < N) atomic_add_f32(&cnt[batch[i]], 1.0f);
}

__global__ void pool_kernel(const float* __restrict__ h, const int* __restrict__ batch,
                            float* __restrict__ out, int N, int C) {
    const int c4 = C >> 2;
    long long tid = (long long)blockIdx.x * blockDim.x + threadIdx.x;
    if (tid >= (long long)N * c4) return;
    int i = (int)(tid / c4);
    int c = (int)(tid % c4) * 4;
    int g = batch[i];
    const float4 v = *(const float4*)(h + (size_t)i * C + c);
    float* o = out + (size_t)g * C + c;
    atomic_add_f32(o + 0, v.x);
    atomic_add_f32(o + 1, v.y);
    atomic_add_f32(o + 2, v.z);
    atomic_add_f32(o + 3, v.w);
}

__global__ void divide_kernel(float* __restrict__ out, const float* __restrict__ cnt,
                              int G, int C) {
    int tid = blockIdx.x * blockDim.x + threadIdx.x;
    if (tid >= G * C) return;
    int g = tid / C;
    out[tid] = out[tid] / fmaxf(cnt[g], 1.0f);
}

// ---------------------------------------------------------------------------
// launch
// ---------------------------------------------------------------------------
static inline int blocks_for(long long n, int t) { return (int)((n + t - 1) / t); }

extern "C" void kernel_launch(void* const* d_in, const int* in_sizes, int n_in,
                              void* d_out, int out_size, void* d_ws, size_t ws_size,
                              hipStream_t stream) {
    const float* x  = (const float*)d_in[0];
    const float* W1 = (const float*)d_in[1];
    const float* b1 = (const float*)d_in[2];
    const float* W2 = (const float*)d_in[3];
    const float* b2 = (const float*)d_in[4];
    const float* W3 = (const float*)d_in[5];
    const float* b3 = (const float*)d_in[6];
    const int* edge_index = (const int*)d_in[7];
    const int* batch      = (const int*)d_in[8];

    const int C_IN = 128, C_HID = 256, C_OUT = 128;
    const int N = in_sizes[0] / C_IN;
    const int E = in_sizes[7] / 2;
    const int G = out_size / C_OUT;
    const int* src = edge_index;        // row 0 of (2,E)
    const int* dst = edge_index + E;    // row 1 of (2,E)

    // workspace carve-up (256B aligned)
    char* ws = (char*)d_ws;
    size_t off = 0;
    auto carve = [&](size_t bytes) -> void* {
        void* p = ws + off;
        off = (off + bytes + 255) & ~(size_t)255;
        return p;
    };
    float*  dinv = (float*)carve((size_t)N * 4);            // deg -> dinv in place
    float*  cnt  = (float*)carve((size_t)G * 4);
    __bf16* Wp1  = (__bf16*)carve((size_t)C_IN  * C_HID * 2);
    __bf16* Wp2  = (__bf16*)carve((size_t)C_HID * C_HID * 2);
    __bf16* Wp3  = (__bf16*)carve((size_t)C_HID * C_OUT * 2);
    float*  bufA = (float*)carve((size_t)N * C_HID * 4);    // lin outputs
    float*  bufB = (float*)carve((size_t)N * C_HID * 4);    // agg / h outputs
    float*  out  = (float*)d_out;

    const int T = 256;

    // pack weights to bf16 B-fragment layout (tiny)
    pack_W_kernel<<<blocks_for(C_IN  * C_HID, T), T, 0, stream>>>(W1, Wp1, C_IN,  C_HID);
    pack_W_kernel<<<blocks_for(C_HID * C_HID, T), T, 0, stream>>>(W2, Wp2, C_HID, C_HID);
    pack_W_kernel<<<blocks_for(C_HID * C_OUT, T), T, 0, stream>>>(W3, Wp3, C_HID, C_OUT);

    // degree + dinv
    zero_f32_kernel<<<blocks_for(N, T), T, 0, stream>>>(dinv, N);
    deg_kernel<<<blocks_for(E, T), T, 0, stream>>>(dst, dinv, E);
    dinv_kernel<<<blocks_for(N, T), T, 0, stream>>>(dinv, N);

    // post-GEMM aggregation path (shared by all layers)
    auto aggregate = [&](const float* b, int Nout, int relu) {
        zero_f32_kernel<<<blocks_for((long long)N * Nout, T), T, 0, stream>>>(
            bufB, (long long)N * Nout);
        scatter_edges<<<blocks_for((long long)E * (Nout / 4), T), T, 0, stream>>>(
            bufA, dinv, src, dst, bufB, E, Nout);
        fuse_kernel<<<blocks_for((long long)N * Nout, T), T, 0, stream>>>(
            bufB, bufA, dinv, b, N, Nout, relu);
    };
    auto gemm_blocks = [&](int Nout) {
        long long waves = (long long)((N + 15) / 16) * (Nout / 64);
        return blocks_for(waves, 8);
    };

    // layer 1: x(N x 128) @ W1 -> 256
    gemm_wmma_bf16<128, 256><<<gemm_blocks(256), 256, 0, stream>>>(x, Wp1, bufA, N);
    aggregate(b1, C_HID, 1);                               // h1 in bufB
    // layer 2: h1(N x 256) @ W2 -> 256
    gemm_wmma_bf16<256, 256><<<gemm_blocks(256), 256, 0, stream>>>(bufB, Wp2, bufA, N);
    aggregate(b2, C_HID, 1);                               // h2 in bufB
    // layer 3: h2(N x 256) @ W3 -> 128
    gemm_wmma_bf16<256, 128><<<gemm_blocks(128), 256, 0, stream>>>(bufB, Wp3, bufA, N);
    aggregate(b3, C_OUT, 0);                               // h3 in bufB (N x 128)

    // global mean pool
    zero_f32_kernel<<<blocks_for(G * C_OUT, T), T, 0, stream>>>(out, (long long)G * C_OUT);
    zero_f32_kernel<<<blocks_for(G, T), T, 0, stream>>>(cnt, G);
    cnt_kernel<<<blocks_for(N, T), T, 0, stream>>>(batch, cnt, N);
    pool_kernel<<<blocks_for((long long)N * (C_OUT / 4), T), T, 0, stream>>>(bufB, batch, out, N, C_OUT);
    divide_kernel<<<blocks_for(G * C_OUT, T), T, 0, stream>>>(out, cnt, G, C_OUT);
}
